// StructuralContrastiveLoss_40226663694515
// MI455X (gfx1250) — matrix operand order repaired and checked by reading, not compiled, not run on hardware
//
#include <hip/hip_runtime.h>
#include <hip/hip_bf16.h>

// ---------------------------------------------------------------------------
// StructuralContrastiveLoss for MI455X (gfx1250, wave32, WMMA)
//
//   dist[i,j] = (sq[i] + sq[j] - 2*gram[i,j]) / 16384 over concatenated
//   K = 3*16384 features; contrastive hinge reduction over i<j.
//
//   gram via v_wmma_f32_16x16x32_bf16 with split-bf16 (hi/lo):
//   x.y ~= hi.hi + hi.lo + lo.hi  (fp32-grade accuracy).
//   LDS-staged, double-buffered, split-K x8; wave tile 32x64 (24 WMMA/iter).
// ---------------------------------------------------------------------------

typedef __attribute__((ext_vector_type(16))) __bf16 v16bf;
typedef __attribute__((ext_vector_type(8)))  float  v8f;

#define BATCH    512
#define FIELD_K  16384
#define NFIELD   3
#define TOTAL_K  (NFIELD * FIELD_K)          // 49152
#define KCHUNKS  (TOTAL_K / 32)              // 1536 WMMA K-steps
#define NSPLIT   8                           // split-K factor
#define KPER     (KCHUNKS / NSPLIT)          // 192 chunks per split
#define NPAIRS   (BATCH * (BATCH - 1) / 2)   // 130816
#define ROWP     40                          // LDS row stride in bf16 (80 B)

union Frag {
    unsigned int u[8];
    v16bf        v;
};

// split two fp32 into packed bf16 hi-pair / lo-pair:
//   hi = truncate-to-bf16 (enables v_fma_mix_f32_bf16 residual fusion)
//   pack = single v_perm_b32
__device__ __forceinline__ void split2(float x0, float x1,
                                       unsigned int& h, unsigned int& l) {
    unsigned int b0 = __float_as_uint(x0), b1 = __float_as_uint(x1);
    h = __builtin_amdgcn_perm(b1, b0, 0x07060302u);
    float r0 = x0 - __uint_as_float(b0 & 0xFFFF0000u);
    float r1 = x1 - __uint_as_float(b1 & 0xFFFF0000u);
    l = __builtin_amdgcn_perm(__float_as_uint(r1), __float_as_uint(r0),
                              0x07060302u);
}

// fragment gather: two 16B LDS loads (ds_load_b128)
__device__ __forceinline__ v16bf read_frag(const unsigned short* p, int row,
                                           int off0, int off1) {
    Frag f;
    uint4 q0 = *reinterpret_cast<const uint4*>(p + row * ROWP + off0);
    uint4 q1 = *reinterpret_cast<const uint4*>(p + row * ROWP + off1);
    f.u[0] = q0.x; f.u[1] = q0.y; f.u[2] = q0.z; f.u[3] = q0.w;
    f.u[4] = q1.x; f.u[5] = q1.y; f.u[6] = q1.z; f.u[7] = q1.w;
    return f.v;
}

// ---------------------------------------------------------------------------
// Kernel 1: per-row sum of squares (exact fp32)
// ---------------------------------------------------------------------------
__global__ __launch_bounds__(256) void rowsumsq_kernel(
    const float* __restrict__ e, const float* __restrict__ a,
    const float* __restrict__ c, float* __restrict__ sq) {
    const int b = blockIdx.x;
    const int t = threadIdx.x;
    const float* srcs[NFIELD] = {e, a, c};
    float acc = 0.0f;
#pragma unroll
    for (int f = 0; f < NFIELD; ++f) {
        const float4* p =
            reinterpret_cast<const float4*>(srcs[f] + (size_t)b * FIELD_K);
        for (int i = t; i < FIELD_K / 4; i += 256) {
            float4 v = p[i];
            acc += v.x * v.x + v.y * v.y + v.z * v.z + v.w * v.w;
        }
    }
    __shared__ float red[256];
    red[t] = acc;
    __syncthreads();
    for (int s = 128; s > 0; s >>= 1) {
        if (t < s) red[t] += red[t + s];
        __syncthreads();
    }
    if (t == 0) sq[b] = red[0];
}

// ---------------------------------------------------------------------------
// Kernel 2: gram slab = X * X^T (partial over this split's K range)
// Block 128 thr = 4 waves; block tile 128x64; wave tile 32x64 (2x4 accums,
// 24 WMMA per K-step). Double-buffered LDS of packed bf16 hi/lo panels.
// ---------------------------------------------------------------------------
__global__ __launch_bounds__(128, 1) void gram_wmma_kernel(
    const float* __restrict__ e, const float* __restrict__ a,
    const float* __restrict__ c, float* __restrict__ gram) {
    const float* srcs[NFIELD] = {e, a, c};

    const int blockRow = blockIdx.x * 128;
    const int blockCol = blockIdx.y * 64;
    if (blockRow >= blockCol + 64) return;   // loss only consumes i<j

    const int tid   = threadIdx.x;
    const int lane  = tid & 31;
    const int waveM = tid >> 5;           // 4 waves stacked in M

    const int mrow  = lane & 15;          // row/col within 16-wide tile
    const int koffA = (lane >> 4) << 3;   // 0 or 8
    const int koffB = (lane >> 4) << 4;   // 0 or 16

    const int kc0 = blockIdx.z * KPER;
    const int kc1 = kc0 + KPER;

    __shared__ __align__(16) unsigned short sAhi[2][128 * ROWP];
    __shared__ __align__(16) unsigned short sAlo[2][128 * ROWP];
    __shared__ __align__(16) unsigned short sBhi[2][64 * ROWP];
    __shared__ __align__(16) unsigned short sBlo[2][64 * ROWP];

    float4 ra[8], rb[4];

    // ---- coalesced global load of one K-chunk (fp32) into registers
    auto load_regs = [&](int kc) {
        const float* base = srcs[kc >> 9];
        const int    kin  = (kc & 511) << 5;
#pragma unroll
        for (int n = 0; n < 8; ++n) {
            int q = tid + n * 128, row = q >> 3, kq = (q & 7) << 2;
            ra[n] = *reinterpret_cast<const float4*>(
                base + (size_t)(blockRow + row) * FIELD_K + kin + kq);
        }
#pragma unroll
        for (int n = 0; n < 4; ++n) {
            int q = tid + n * 128, row = q >> 3, kq = (q & 7) << 2;
            rb[n] = *reinterpret_cast<const float4*>(
                base + (size_t)(blockCol + row) * FIELD_K + kin + kq);
        }
    };

    // ---- convert to bf16 hi/lo and store into LDS buffer p
    auto store_lds = [&](int p) {
#pragma unroll
        for (int n = 0; n < 8; ++n) {
            int q = tid + n * 128, row = q >> 3, kq = (q & 7) << 2;
            uint2 h, l;
            split2(ra[n].x, ra[n].y, h.x, l.x);
            split2(ra[n].z, ra[n].w, h.y, l.y);
            *reinterpret_cast<uint2*>(&sAhi[p][row * ROWP + kq]) = h;
            *reinterpret_cast<uint2*>(&sAlo[p][row * ROWP + kq]) = l;
        }
#pragma unroll
        for (int n = 0; n < 4; ++n) {
            int q = tid + n * 128, row = q >> 3, kq = (q & 7) << 2;
            uint2 h, l;
            split2(rb[n].x, rb[n].y, h.x, l.x);
            split2(rb[n].z, rb[n].w, h.y, l.y);
            *reinterpret_cast<uint2*>(&sBhi[p][row * ROWP + kq]) = h;
            *reinterpret_cast<uint2*>(&sBlo[p][row * ROWP + kq]) = l;
        }
    };

    v8f acc[2][4] = {};

    load_regs(kc0);
    store_lds(0);
    __syncthreads();

    int p = 0;
    for (int kc = kc0; kc < kc1; ++kc) {
        const bool more = (kc + 1 < kc1);
        if (more) load_regs(kc + 1);   // global loads overlap the WMMAs

        // A fragments for this wave's 32 rows (hi & lo)
        v16bf ahi[2], alo[2];
#pragma unroll
        for (int i = 0; i < 2; ++i) {
            int rA = waveM * 32 + i * 16 + mrow;
            ahi[i] = read_frag(sAhi[p], rA, koffA, 16 + koffA);
            alo[i] = read_frag(sAlo[p], rA, koffA, 16 + koffA);
        }
        // stream B fragments across the 64-wide N extent
#pragma unroll
        for (int j = 0; j < 4; ++j) {
            int rB = j * 16 + mrow;
            v16bf bh = read_frag(sBhi[p], rB, koffA, 16 + koffA);
            v16bf bl = read_frag(sBlo[p], rB, koffA, 16 + koffA);
#pragma unroll
            for (int i = 0; i < 2; ++i) {
                v8f cacc = acc[i][j];
                cacc = __builtin_amdgcn_wmma_f32_16x16x32_bf16(
                    false, ahi[i], false, bh, (short)0, cacc, false, false);
                cacc = __builtin_amdgcn_wmma_f32_16x16x32_bf16(
                    false, ahi[i], false, bl, (short)0, cacc, false, false);
                cacc = __builtin_amdgcn_wmma_f32_16x16x32_bf16(
                    false, alo[i], false, bh, (short)0, cacc, false, false);
                acc[i][j] = cacc;
            }
        }

        if (more) store_lds(p ^ 1);
        __syncthreads();
        p ^= 1;
    }

    // ---- epilogue: C/D layout -> gram slab for this split
    float* gslab = gram + (size_t)blockIdx.z * BATCH * BATCH;
    const int mbase = (lane >> 4) << 3;
    const int ncol  = lane & 15;
#pragma unroll
    for (int i = 0; i < 2; ++i) {
#pragma unroll
        for (int j = 0; j < 4; ++j) {
            const int col = blockCol + j * 16 + ncol;
            const int rw  = blockRow + waveM * 32 + i * 16 + mbase;
#pragma unroll
            for (int r = 0; r < 8; ++r) {
                gslab[(size_t)(rw + r) * BATCH + col] = acc[i][j][r];
            }
        }
    }
}

// ---------------------------------------------------------------------------
// Kernel 3a: per-block partial contrastive sums (sums the NSPLIT slabs)
// ---------------------------------------------------------------------------
__global__ __launch_bounds__(256) void loss_partial_kernel(
    const float* __restrict__ gram, const float* __restrict__ sq,
    const long long* __restrict__ targets, float* __restrict__ partial) {
    const int t   = threadIdx.x;
    const int gid = blockIdx.x * 256 + t;
    float acc = 0.0f;
#pragma unroll
    for (int n = 0; n < 4; ++n) {
        const int idx = gid + n * 65536;
        const int i = idx >> 9;
        const int j = idx & (BATCH - 1);
        if (j > i) {
            float g = 0.0f;
#pragma unroll
            for (int s = 0; s < NSPLIT; ++s)
                g += gram[(size_t)s * BATCH * BATCH + idx];
            float d = (sq[i] + sq[j] - 2.0f * g) * (1.0f / (float)FIELD_K);
            acc += (targets[i] == targets[j]) ? d : fmaxf(1.0f - d, 0.0f);
        }
    }
    __shared__ float red[256];
    red[t] = acc;
    __syncthreads();
    for (int s = 128; s > 0; s >>= 1) {
        if (t < s) red[t] += red[t + s];
        __syncthreads();
    }
    if (t == 0) partial[blockIdx.x] = red[0];
}

// ---------------------------------------------------------------------------
// Kernel 3b: final reduction of 256 partials -> scalar loss
// ---------------------------------------------------------------------------
__global__ __launch_bounds__(256) void loss_final_kernel(
    const float* __restrict__ partial, float* __restrict__ out) {
    const int t = threadIdx.x;
    __shared__ float red[256];
    red[t] = partial[t];
    __syncthreads();
    for (int s = 128; s > 0; s >>= 1) {
        if (t < s) red[t] += red[t + s];
        __syncthreads();
    }
    if (t == 0) out[0] = red[0] / (float)NPAIRS;
}

// ---------------------------------------------------------------------------
extern "C" void kernel_launch(void* const* d_in, const int* in_sizes, int n_in,
                              void* d_out, int out_size, void* d_ws,
                              size_t ws_size, hipStream_t stream) {
    (void)in_sizes; (void)n_in; (void)out_size; (void)ws_size;
    const float*     entropy   = (const float*)d_in[0];
    const float*     alignment = (const float*)d_in[1];
    const float*     curvature = (const float*)d_in[2];
    const long long* targets   = (const long long*)d_in[3];

    float* sq      = (float*)d_ws;       // 512 floats
    float* partial = sq + BATCH;         // 256 floats
    float* gram    = partial + 256;      // NSPLIT * 512*512 floats (8 MB)

    rowsumsq_kernel<<<dim3(BATCH), dim3(256), 0, stream>>>(
        entropy, alignment, curvature, sq);
    gram_wmma_kernel<<<dim3(4, 8, NSPLIT), dim3(128), 0, stream>>>(
        entropy, alignment, curvature, gram);
    loss_partial_kernel<<<dim3(256), dim3(256), 0, stream>>>(
        gram, sq, targets, partial);
    loss_final_kernel<<<dim3(1), dim3(256), 0, stream>>>(
        partial, (float*)d_out);
}
